// HunyuanMoE_78469052498384
// MI455X (gfx1250) — matrix operand order repaired and checked by reading, not compiled, not run on hardware
//
#include <hip/hip_runtime.h>
#include <hip/hip_fp16.h>

typedef __attribute__((ext_vector_type(16))) _Float16 v16h;
typedef __attribute__((ext_vector_type(8)))  float    v8f;
typedef __attribute__((ext_vector_type(2)))  __fp16   v2fp;

#define TOPK   8
#define NEXP   64
#define NVEXP  68          // 64 routed experts + 4 virtual experts carrying the shared MLP
#define FFN    3072
#define HDIM   1024
#define NTOK   512
#define CAP    160
#define MAXMT  10          // ceil(CAP/16)
#define A_STRIDE 40        // halves per A-panel row in LDS (32 data + 8 pad -> 80B rows)

// ---------------- helpers ----------------

__device__ __forceinline__ v8f zero8() {
    v8f z;
#pragma unroll
    for (int i = 0; i < 8; ++i) z[i] = 0.0f;
    return z;
}

// v_cvt_pk_rtz_f16_f32: pack two fp32 into two f16, return raw 32-bit payload
__device__ __forceinline__ unsigned int pk2(float a, float b) {
    union { v2fp h2; unsigned int u; } U;
    U.h2 = __builtin_amdgcn_cvt_pkrtz(a, b);
    return U.u;
}

// Load fp32 [0..7] and [16..23] from global, convert to 16 halves (one B-fragment lane slice).
__device__ __forceinline__ v16h load_b_f32(const float* __restrict__ p) {
    float4 f0 = *(const float4*)(p + 0);
    float4 f1 = *(const float4*)(p + 4);
    float4 f2 = *(const float4*)(p + 16);
    float4 f3 = *(const float4*)(p + 20);
    union { unsigned int u[8]; v16h v; } B;
    B.u[0] = pk2(f0.x, f0.y);
    B.u[1] = pk2(f0.z, f0.w);
    B.u[2] = pk2(f1.x, f1.y);
    B.u[3] = pk2(f1.z, f1.w);
    B.u[4] = pk2(f2.x, f2.y);
    B.u[5] = pk2(f2.z, f2.w);
    B.u[6] = pk2(f3.x, f3.y);
    B.u[7] = pk2(f3.z, f3.w);
    return B.v;
}

// A-fragment: lanes 0-15 -> row, K {0..7,16..23}; lanes 16-31 -> row, K {8..15,24..31}
__device__ __forceinline__ v16h load_a_lds(const _Float16* p) {
    union { uint4 u[2]; v16h v; } U;
    U.u[0] = *(const uint4*)(p);        // halves [hs*8 .. hs*8+7]
    U.u[1] = *(const uint4*)(p + 16);   // halves [16+hs*8 .. 16+hs*8+7]
    return U.v;
}

// Stage gathered fp32 token panel -> f16 LDS panel (MT*16 rows x 32 halves)
template <int MT>
__device__ __forceinline__ void stage_a_f32(const float* __restrict__ X,
                                            const int* sTok, _Float16* buf,
                                            int k0, int tid) {
#pragma unroll
    for (int it = 0; it < MT * 2; ++it) {       // MT*256 half-pairs / 128 threads
        int idx = tid + it * 128;
        int r = idx >> 4, k2 = (idx & 15) * 2;
        int t = sTok[r];
        float2 f = *(const float2*)(X + (size_t)t * HDIM + k0 + k2);
        *(unsigned int*)&buf[r * A_STRIDE + k2] = pk2(f.x, f.y);
    }
}

// Stage contiguous f16 act panel -> LDS panel
template <int MT>
__device__ __forceinline__ void stage_a_f16(const _Float16* __restrict__ arow,
                                            _Float16* buf, int k0, int tid) {
#pragma unroll
    for (int it = 0; it < MT * 2; ++it) {
        int idx = tid + it * 128;
        int r = idx >> 4, k2 = (idx & 15) * 2;
        *(unsigned int*)&buf[r * A_STRIDE + k2] =
            *(const unsigned int*)(arow + (size_t)r * FFN + k0 + k2);
    }
}

#define DISPATCH_MT(fn, ...)                       \
    switch (mtiles) {                              \
        case 1:  fn<1>(__VA_ARGS__);  break;       \
        case 2:  fn<2>(__VA_ARGS__);  break;       \
        case 3:  fn<3>(__VA_ARGS__);  break;       \
        case 4:  fn<4>(__VA_ARGS__);  break;       \
        case 5:  fn<5>(__VA_ARGS__);  break;       \
        case 6:  fn<6>(__VA_ARGS__);  break;       \
        case 7:  fn<7>(__VA_ARGS__);  break;       \
        case 8:  fn<8>(__VA_ARGS__);  break;       \
        case 9:  fn<9>(__VA_ARGS__);  break;       \
        default: fn<10>(__VA_ARGS__); break;       \
    }

// ---------------- kernel 1: init ----------------

__global__ void moe_init(float* __restrict__ out, int* __restrict__ counts,
                         int* __restrict__ slot_tok, float* __restrict__ slot_w) {
    int i = blockIdx.x * blockDim.x + threadIdx.x;
    for (int idx = i; idx < NTOK * HDIM; idx += gridDim.x * blockDim.x) out[idx] = 0.0f;
    if (i < NVEXP) counts[i] = (i < NEXP) ? 0 : 128;   // virtual experts: 128 tokens each
    if (i < NTOK) {
        int ve = NEXP + (i >> 7);
        int c  = i & 127;
        slot_tok[ve * CAP + c] = i;
        slot_w[ve * CAP + c]   = 1.0f;
    }
}

// ---------------- kernel 2: gating + dispatch ----------------

__global__ void __launch_bounds__(64) moe_gate(const float* __restrict__ X,
                                               const float* __restrict__ WG,
                                               int* __restrict__ counts,
                                               int* __restrict__ slot_tok,
                                               float* __restrict__ slot_w) {
    __shared__ float xs[8 * HDIM];     // 32 KB
    __shared__ float lg[8 * NEXP];     // 2 KB
    const int tid = threadIdx.x;
    const int t0  = blockIdx.x * 8;

    for (int i = tid; i < 8 * HDIM; i += 64) xs[i] = X[(size_t)t0 * HDIM + i];
    __syncthreads();

    float acc[8];
#pragma unroll
    for (int t = 0; t < 8; ++t) acc[t] = 0.0f;
    const float* wrow = WG + (size_t)tid * HDIM;
    for (int k = 0; k < HDIM; ++k) {
        float wv = wrow[k];
#pragma unroll
        for (int t = 0; t < 8; ++t) acc[t] += xs[t * HDIM + k] * wv;  // LDS broadcast
    }
#pragma unroll
    for (int t = 0; t < 8; ++t) lg[t * NEXP + tid] = acc[t];
    __syncthreads();

    if (tid < 8) {
        float* row = &lg[tid * NEXP];
        float vals[TOPK]; int ids[TOPK];
        for (int s = 0; s < TOPK; ++s) {
            float best = -1e30f; int bi = 0;
            for (int e = 0; e < NEXP; ++e) { float v = row[e]; if (v > best) { best = v; bi = e; } }
            vals[s] = best; ids[s] = bi; row[bi] = -1e30f;
        }
        float m = vals[0], sum = 0.0f, w8[TOPK];
        for (int s = 0; s < TOPK; ++s) { w8[s] = __expf(vals[s] - m); sum += w8[s]; }
        float inv = 1.0f / sum;
        int tg = t0 + tid;
        for (int s = 0; s < TOPK; ++s) {
            int e = ids[s];
            int r = atomicAdd(&counts[e], 1);
            if (r < CAP) { slot_tok[e * CAP + r] = tg; slot_w[e * CAP + r] = w8[s] * inv; }
        }
    }
}

// ---------------- kernel 3: grouped GEMM1 + fused SwiGLU ----------------

template <int MT>
__device__ __forceinline__ void w1_body(const float* __restrict__ X,
                                        const float* __restrict__ W,
                                        const int* sTok,
                                        _Float16* __restrict__ act,
                                        _Float16* bufA, _Float16* bufB,
                                        int e, int j0, int cnt, int tid) {
    const int lane = tid & 31, wave = tid >> 5;
    const int hs = lane >> 4, ln = lane & 15;

    const float* b1p = W + (size_t)(j0 + wave * 16 + ln) * HDIM + hs * 8;
    const float* b2p = W + (size_t)(FFN + j0 + wave * 16 + ln) * HDIM + hs * 8;

    v8f acc1[MT], acc2[MT];
#pragma unroll
    for (int i = 0; i < MT; ++i) { acc1[i] = zero8(); acc2[i] = zero8(); }

    stage_a_f32<MT>(X, sTok, bufA, 0, tid);
    _Float16* cur = bufA;
    _Float16* nxt = bufB;

    for (int k0 = 0; k0 < HDIM; k0 += 32) {
        __syncthreads();
        __builtin_prefetch(b1p + k0 + 32, 0, 1);    // next 128B of weight rows
        __builtin_prefetch(b2p + k0 + 32, 0, 1);
        if (k0 + 32 < HDIM) stage_a_f32<MT>(X, sTok, nxt, k0 + 32, tid);
        v16h b1 = load_b_f32(b1p + k0);
        v16h b2 = load_b_f32(b2p + k0);
#pragma unroll
        for (int mt = 0; mt < MT; ++mt) {
            v16h a = load_a_lds(&cur[(mt * 16 + ln) * A_STRIDE + hs * 8]);
            acc1[mt] = __builtin_amdgcn_wmma_f32_16x16x32_f16(false, a, false, b1,
                                                              (short)0, acc1[mt], false, false);
            acc2[mt] = __builtin_amdgcn_wmma_f32_16x16x32_f16(false, a, false, b2,
                                                              (short)0, acc2[mt], false, false);
        }
        _Float16* t = cur; cur = nxt; nxt = t;
    }

    const int col = j0 + wave * 16 + ln;
#pragma unroll
    for (int mt = 0; mt < MT; ++mt) {
#pragma unroll
        for (int r = 0; r < 8; ++r) {
            int m = mt * 16 + hs * 8 + r;
            if (m < cnt) {
                float x1 = acc1[mt][r];
                float z  = acc2[mt][r];
                float a  = x1 * (z * (1.0f / (1.0f + __expf(-z))));   // x1 * silu(x2)
                act[(size_t)(e * CAP + m) * FFN + col] = (_Float16)a;
            }
        }
    }
}

__global__ void __launch_bounds__(128) moe_w1(const float* __restrict__ X,
                                              const float* __restrict__ shared_w1,
                                              const float* __restrict__ expert_w1,
                                              const int* __restrict__ counts,
                                              const int* __restrict__ slot_tok,
                                              _Float16* __restrict__ act) {
    const int e = blockIdx.x;
    int cnt = counts[e];
    if (cnt <= 0) return;
    if (cnt > CAP) cnt = CAP;
    const int mtiles = (cnt + 15) >> 4;
    const int j0     = blockIdx.y * 64;

    const float* W = (e < NEXP) ? (expert_w1 + (size_t)e * (2 * FFN) * HDIM) : shared_w1;

    __shared__ __align__(16) _Float16 aLds[2][CAP * A_STRIDE];
    __shared__ int sTok[CAP];

    const int tid = threadIdx.x;
    const int base = e * CAP;
    for (int i = tid; i < CAP; i += 128) sTok[i] = (i < cnt) ? slot_tok[base + i] : 0;
    __syncthreads();

    DISPATCH_MT(w1_body, X, W, sTok, act, aLds[0], aLds[1], e, j0, cnt, tid);
}

// ---------------- kernel 4: grouped GEMM2 + weighted scatter-combine ----------------

template <int MT>
__device__ __forceinline__ void w2_body(const _Float16* __restrict__ arow,
                                        const float* __restrict__ W,
                                        const int* sTok, const float* sWgt,
                                        float* __restrict__ out,
                                        _Float16* bufA, _Float16* bufB,
                                        int n0, int cnt, int tid) {
    const int lane = tid & 31, wave = tid >> 5;
    const int hs = lane >> 4, ln = lane & 15;
    const float* bp = W + (size_t)(n0 + wave * 16 + ln) * FFN + hs * 8;

    v8f acc[MT];
#pragma unroll
    for (int i = 0; i < MT; ++i) acc[i] = zero8();

    stage_a_f16<MT>(arow, bufA, 0, tid);
    _Float16* cur = bufA;
    _Float16* nxt = bufB;

    for (int k0 = 0; k0 < FFN; k0 += 32) {
        __syncthreads();
        __builtin_prefetch(bp + k0 + 32, 0, 1);
        if (k0 + 32 < FFN) stage_a_f16<MT>(arow, nxt, k0 + 32, tid);
        v16h b = load_b_f32(bp + k0);
#pragma unroll
        for (int mt = 0; mt < MT; ++mt) {
            v16h a = load_a_lds(&cur[(mt * 16 + ln) * A_STRIDE + hs * 8]);
            acc[mt] = __builtin_amdgcn_wmma_f32_16x16x32_f16(false, a, false, b,
                                                             (short)0, acc[mt], false, false);
        }
        _Float16* t = cur; cur = nxt; nxt = t;
    }

    const int col = n0 + wave * 16 + ln;
#pragma unroll
    for (int mt = 0; mt < MT; ++mt) {
#pragma unroll
        for (int r = 0; r < 8; ++r) {
            int m = mt * 16 + hs * 8 + r;
            if (m < cnt) {
                float v = acc[mt][r] * sWgt[m];
                atomicAdd(&out[(size_t)sTok[m] * HDIM + col], v);
            }
        }
    }
}

__global__ void __launch_bounds__(128) moe_w2(const _Float16* __restrict__ act,
                                              const float* __restrict__ shared_w2,
                                              const float* __restrict__ expert_w2,
                                              const int* __restrict__ counts,
                                              const int* __restrict__ slot_tok,
                                              const float* __restrict__ slot_w,
                                              float* __restrict__ out) {
    const int e = blockIdx.x;
    int cnt = counts[e];
    if (cnt <= 0) return;
    if (cnt > CAP) cnt = CAP;
    const int mtiles = (cnt + 15) >> 4;
    const int n0     = blockIdx.y * 64;

    const float* W = (e < NEXP) ? (expert_w2 + (size_t)e * HDIM * FFN) : shared_w2;
    const _Float16* arow = act + (size_t)e * CAP * FFN;

    __shared__ __align__(16) _Float16 aLds[2][CAP * A_STRIDE];
    __shared__ int   sTok[CAP];
    __shared__ float sWgt[CAP];

    const int tid = threadIdx.x;
    const int base = e * CAP;
    for (int i = tid; i < CAP; i += 128) {
        sTok[i] = (i < cnt) ? slot_tok[base + i] : 0;
        sWgt[i] = (i < cnt) ? slot_w[base + i] : 0.0f;
    }
    __syncthreads();

    DISPATCH_MT(w2_body, arow, W, sTok, sWgt, out, aLds[0], aLds[1], n0, cnt, tid);
}

// ---------------- launcher ----------------

extern "C" void kernel_launch(void* const* d_in, const int* in_sizes, int n_in,
                              void* d_out, int out_size, void* d_ws, size_t ws_size,
                              hipStream_t stream) {
    const float* X   = (const float*)d_in[0];   // [1,512,1024]
    const float* WG  = (const float*)d_in[1];   // [64,1024]
    const float* SW1 = (const float*)d_in[2];   // [6144,1024]
    const float* SW2 = (const float*)d_in[3];   // [1024,3072]
    const float* EW1 = (const float*)d_in[4];   // [64,6144,1024]
    const float* EW2 = (const float*)d_in[5];   // [64,1024,3072]
    float* out = (float*)d_out;

    char* ws = (char*)d_ws;
    const size_t oTok = 512;
    const size_t oWgt = oTok + (size_t)NVEXP * CAP * sizeof(int);
    const size_t oAct = oWgt + (size_t)NVEXP * CAP * sizeof(float);  // 256B-aligned
    int*      counts   = (int*)ws;
    int*      slot_tok = (int*)(ws + oTok);
    float*    slot_w   = (float*)(ws + oWgt);
    _Float16* act      = (_Float16*)(ws + oAct);   // NVEXP*CAP*FFN halves (~67 MB)

    moe_init<<<dim3(512), 256, 0, stream>>>(out, counts, slot_tok, slot_w);
    moe_gate<<<dim3(NTOK / 8), 64, 0, stream>>>(X, WG, counts, slot_tok, slot_w);
    moe_w1<<<dim3(NVEXP, FFN / 64), 128, 0, stream>>>(X, SW1, EW1, counts, slot_tok, act);
    moe_w2<<<dim3(NVEXP, HDIM / 64), 128, 0, stream>>>(act, SW2, EW2, counts, slot_tok, slot_w, out);
}
